// NueralContJumpProcess_76355928589009
// MI455X (gfx1250) — matrix operand order repaired and checked by reading, not compiled, not run on hardware
//
#include <hip/hip_runtime.h>
#include <hip/hip_bf16.h>
#include <math.h>

// ---------------------------------------------------------------------------
// Neural continuous jump process on MI455X (gfx1250).
// GEMMs: v_wmma_f32_16x16x32_bf16, B staged to LDS via async copies
// (global_load_async_to_lds_b128, ASYNCcnt double-buffering), A from global.
// State h kept in f32 (RK4 accumulation); bf16 shadows feed the matrix units.
// Epilogue tanh uses the gfx1250 V_TANH_F32 transcendental when available.
// ---------------------------------------------------------------------------

typedef __attribute__((ext_vector_type(16))) __bf16       v16bf;
typedef __attribute__((ext_vector_type(8)))  float        v8f;
typedef __attribute__((ext_vector_type(4)))  unsigned int u32x4;

constexpr int N_   = 128;
constexpr int T_   = 16;
constexpr int D_   = 25;
constexpr int H_   = 256;
constexpr int HC_  = 512;
constexpr int HI_  = 128;
constexpr int E_   = 128;
constexpr int HP_  = 256;
constexpr int KSUB = 8;
constexpr int L_   = D_ * (D_ - 1) + 8;   // 608
constexpr int DH_  = D_ * H_;             // 6400
constexpr int ND_  = N_ * D_;             // 3200
constexpr int NDH_ = N_ * D_ * H_;        // 819200

union AFrag { v16bf v; unsigned int u[8]; };
union BFrag { v16bf v; u32x4 q[2]; };

__device__ __forceinline__ float fast_tanh(float x) {
#if __has_builtin(__builtin_amdgcn_tanhf)
  return __builtin_amdgcn_tanhf(x);      // V_TANH_F32 (TRANS32)
#elif __has_builtin(__builtin_amdgcn_tanh_f32)
  return __builtin_amdgcn_tanh_f32(x);
#else
  return tanhf(x);
#endif
}

// ---- async copy of one 64(K) x NT*16(N) B chunk into LDS -------------------
// 128 threads; each thread issues NT b128 copies. Padded row = NT*32+16 B.
template <int NT>
__device__ __forceinline__ void async_issue_chunk(
    const __bf16* __restrict__ B, int ldb, int n0, int kk, char* buf) {
  constexpr int SEGROW = NT * 2;        // 16B segments per row
  constexpr int ROWB   = NT * 32 + 16;  // padded LDS row bytes
  const int tid = threadIdx.x;
  const __bf16* gb = B + (size_t)kk * ldb + n0;
#pragma unroll
  for (int s = 0; s < NT; ++s) {
    const int q  = tid + s * 128;       // 0 .. 128*NT-1 (rows 0..63)
    const int r  = q / SEGROW;
    const int cs = q - r * SEGROW;
    const unsigned lds  = (unsigned)(size_t)(buf + r * ROWB + cs * 16);
    const unsigned voff = (unsigned)(((unsigned)r * (unsigned)ldb + cs * 8u) * 2u);
    asm volatile("global_load_async_to_lds_b128 %0, %1, %2"
                 :: "v"(lds), "v"(voff), "s"(gb) : "memory");
  }
}

// ---- WMMA core: one wave accumulates a 16(M) x NT*16(N) strip --------------
// B flows through double-buffered LDS chunks (ASYNCcnt pipelined).
template <int NT>
__device__ __forceinline__ void wmma_strip_accum(
    const __bf16* __restrict__ A, const __bf16* __restrict__ B,
    int Kk, int ldb, int mrow, int kbase, int lane, int n0,
    char* sB0, char* sB1, v8f acc[NT]) {
  constexpr int ROWB = NT * 32 + 16;
  const int nch = Kk >> 6;              // K chunks of 64 (all K are mult of 64)
  char* bufs[2] = {sB0, sB1};
  async_issue_chunk<NT>(B, ldb, n0, 0, sB0);
  if (nch > 1) async_issue_chunk<NT>(B, ldb, n0, 64, sB1);
  for (int ic = 0; ic < nch; ++ic) {
    if (ic + 1 < nch) {   // one chunk (NT wave-instructions) may stay in flight
      if constexpr (NT == 4) asm volatile("s_wait_asynccnt 0x4" ::: "memory");
      else                   asm volatile("s_wait_asynccnt 0x2" ::: "memory");
    } else {
      asm volatile("s_wait_asynccnt 0x0" ::: "memory");
    }
    __syncthreads();
    const char* bb = bufs[ic & 1];
    const int kk = ic * 64;
#pragma unroll
    for (int j = 0; j < 2; ++j) {
      // A fragment: 16-bit A 16x32 layout -> two contiguous 16B runs.
      AFrag af;
      const __bf16* ap = A + (size_t)mrow * Kk + kk + j * 32 + kbase;
#pragma unroll
      for (int p = 0; p < 8; ++p) {
        const int k = ((p & 4) << 2) | ((p & 3) << 1);
        af.u[p] = *(const unsigned int*)(ap + k);
      }
      // B fragments from LDS: lane holds column n=lane&15, k in A's table order.
      const char* brow = bb + (j * 32 + ((lane >> 4) << 3)) * ROWB + (lane & 15) * 2;
#pragma unroll
      for (int c = 0; c < NT; ++c) {
        BFrag bf;
#pragma unroll
        for (int e = 0; e < 16; ++e)
          bf.v[e] = *(const __bf16*)(brow + ((e & 7) + ((e >> 3) << 4)) * ROWB + c * 32);
        acc[c] = __builtin_amdgcn_wmma_f32_16x16x32_bf16(
            false, af.v, false, bf.v, (short)0, acc[c], false, false);
      }
    }
    __syncthreads();
    if (ic + 2 < nch)
      async_issue_chunk<NT>(B, ldb, n0, (ic + 2) * 64, bufs[ic & 1]);
  }
}

// Per-wave tile coordinates (4 waves of 16 rows per 128-thread block).
struct TileCoord {
  int lane, rowTile, mrow, kbase, gmBase, gn0;
};
__device__ __forceinline__ TileCoord tile_coord() {
  TileCoord tc;
  tc.lane    = threadIdx.x & 31;
  const int wv = threadIdx.x >> 5;
  tc.rowTile = blockIdx.x * 64 + wv * 16;
  tc.mrow    = tc.rowTile + (tc.lane & 15);
  tc.kbase   = (tc.lane >> 4) << 3;
  // C/D layout: VGPR r, lane l -> M = r + (l>=16 ? 8 : 0), N = l & 15.
  tc.gmBase  = tc.rowTile + ((tc.lane >> 4) << 3);
  tc.gn0     = tc.lane & 15;
  return tc;
}

// ---- Generic GEMM: C = act(A(MxK) * B(KxN) + bias), f32 and/or bf16 stores -
// Grid: x = M/64, y = strips of NT*16 cols starting at n0base.
template <int NT, int ACT, bool WF, bool WB>
__global__ __launch_bounds__(128) void gemm_bf16_kernel(
    const __bf16* __restrict__ A, const __bf16* __restrict__ B,
    const float* __restrict__ bias, float* __restrict__ Cf,
    __bf16* __restrict__ Cb, int Kk, int ldb, int ldc, int n0base) {
  __shared__ alignas(16) char sB[2][64 * (NT * 32 + 16)];
  const TileCoord tc = tile_coord();
  const int n0 = n0base + blockIdx.y * (NT * 16);
  v8f acc[NT];
#pragma unroll
  for (int c = 0; c < NT; ++c) acc[c] = (v8f){0.f,0.f,0.f,0.f,0.f,0.f,0.f,0.f};
  wmma_strip_accum<NT>(A, B, Kk, ldb, tc.mrow, tc.kbase, tc.lane, n0,
                       sB[0], sB[1], acc);
#pragma unroll
  for (int c = 0; c < NT; ++c) {
    const int gn = n0 + c * 16 + tc.gn0;
    const float bv = bias[gn];
#pragma unroll
    for (int r = 0; r < 8; ++r) {
      const int gm = tc.gmBase + r;
      float v = acc[c][r] + bv;
      if (ACT == 1) v = fast_tanh(v);
      if constexpr (WF) Cf[(size_t)gm * ldc + gn] = v;
      if constexpr (WB) Cb[(size_t)gm * ldc + gn] = (__bf16)v;
    }
  }
}

// ---- Jump GEMM: h += tanh(tmp@wd + obs*wo + bd); refresh bf16 h ------------
__global__ __launch_bounds__(128) void gemm_jump_kernel(
    const __bf16* __restrict__ A /*tmp (ND x H)*/,
    const __bf16* __restrict__ B /*wd (H x H)*/,
    const float* __restrict__ bd, const float* __restrict__ wo,
    const float* __restrict__ ED /*(N,T,D)*/, int t,
    float* __restrict__ hf, __bf16* __restrict__ hb) {
  __shared__ alignas(16) char sB[2][64 * 144];
  const TileCoord tc = tile_coord();
  const int n0 = blockIdx.y * 64;
  v8f acc[4];
#pragma unroll
  for (int c = 0; c < 4; ++c) acc[c] = (v8f){0.f,0.f,0.f,0.f,0.f,0.f,0.f,0.f};
  wmma_strip_accum<4>(A, B, H_, H_, tc.mrow, tc.kbase, tc.lane, n0,
                      sB[0], sB[1], acc);
#pragma unroll
  for (int c = 0; c < 4; ++c) {
    const int gn = n0 + c * 16 + tc.gn0;
    const float bv = bd[gn], wov = wo[gn];
#pragma unroll
    for (int r = 0; r < 8; ++r) {
      const int gm = tc.gmBase + r;        // gm = n*D + i
      const int n  = gm / D_;
      const int i  = gm - n * D_;
      const float obs = ED[((size_t)n * T_ + t) * D_ + i];
      const float v   = fast_tanh(acc[c][r] + bv + obs * wov);
      const size_t idx = (size_t)gm * H_ + gn;
      const float hn = hf[idx] + v;
      hf[idx] = hn;
      hb[idx] = (__bf16)hn;
    }
  }
}

// ---- ODE stage GEMM: k = g1@wc2 + bc2 ; x = bf16(h + coef*s[n]*k) ----------
__global__ __launch_bounds__(128) void gemm_cfn2_stage_kernel(
    const __bf16* __restrict__ A /*g1 (ND x HC)*/,
    const __bf16* __restrict__ B /*wc2 (HC x H)*/,
    const float* __restrict__ bc2, const float* __restrict__ hf,
    const float* __restrict__ s, float coef,
    float* __restrict__ kout, __bf16* __restrict__ xb) {
  __shared__ alignas(16) char sB[2][64 * 144];
  const TileCoord tc = tile_coord();
  const int n0 = blockIdx.y * 64;
  v8f acc[4];
#pragma unroll
  for (int c = 0; c < 4; ++c) acc[c] = (v8f){0.f,0.f,0.f,0.f,0.f,0.f,0.f,0.f};
  wmma_strip_accum<4>(A, B, HC_, H_, tc.mrow, tc.kbase, tc.lane, n0,
                      sB[0], sB[1], acc);
#pragma unroll
  for (int c = 0; c < 4; ++c) {
    const int gn = n0 + c * 16 + tc.gn0;
    const float bv = bc2[gn];
#pragma unroll
    for (int r = 0; r < 8; ++r) {
      const int gm = tc.gmBase + r;        // gm = n*D + i
      const int n  = gm / D_;
      const float kv = acc[c][r] + bv;
      const size_t idx = (size_t)gm * H_ + gn;
      kout[idx] = kv;
      xb[idx] = (__bf16)(hf[idx] + coef * s[n] * kv);
    }
  }
}

// ---- ODE final GEMM: k4 = acc+bc2 ; h += (s/6)(k1+2k2+2k3+k4) --------------
__global__ __launch_bounds__(128) void gemm_cfn2_final_kernel(
    const __bf16* __restrict__ A /*g1*/, const __bf16* __restrict__ B /*wc2*/,
    const float* __restrict__ bc2,
    const float* __restrict__ k1, const float* __restrict__ k2,
    const float* __restrict__ k3, const float* __restrict__ s,
    float* __restrict__ hf, __bf16* __restrict__ hb) {
  __shared__ alignas(16) char sB[2][64 * 144];
  const TileCoord tc = tile_coord();
  const int n0 = blockIdx.y * 64;
  v8f acc[4];
#pragma unroll
  for (int c = 0; c < 4; ++c) acc[c] = (v8f){0.f,0.f,0.f,0.f,0.f,0.f,0.f,0.f};
  wmma_strip_accum<4>(A, B, HC_, H_, tc.mrow, tc.kbase, tc.lane, n0,
                      sB[0], sB[1], acc);
#pragma unroll
  for (int c = 0; c < 4; ++c) {
    const int gn = n0 + c * 16 + tc.gn0;
    const float bv = bc2[gn];
#pragma unroll
    for (int r = 0; r < 8; ++r) {
      const int gm = tc.gmBase + r;
      const int n  = gm / D_;
      const float k4v = acc[c][r] + bv;
      const size_t idx = (size_t)gm * H_ + gn;
      const float v = hf[idx] +
          (s[n] / 6.0f) * (k1[idx] + 2.0f * k2[idx] + 2.0f * k3[idx] + k4v);
      hf[idx] = v;
      hb[idx] = (__bf16)v;
    }
  }
}

// ---- small helpers ---------------------------------------------------------
__global__ void convert_bf16_kernel(const float* __restrict__ s,
                                    __bf16* __restrict__ d, int n) {
  int i = blockIdx.x * 256 + threadIdx.x;
  if (i < n) d[i] = (__bf16)s[i];
}

__global__ void hinit_kernel(const float* __restrict__ h0,
                             float* __restrict__ hf, __bf16* __restrict__ hb) {
  int i = blockIdx.x * 256 + threadIdx.x;  // grid covers NDH_ exactly
  float v = h0[i % DH_];
  hf[i] = v;
  hb[i] = (__bf16)v;
}

__global__ void dt_kernel(const float* __restrict__ ET,
                          const float* __restrict__ t0v,
                          float* __restrict__ s, int t) {
  int n = threadIdx.x;
  if (n < N_) {
    float tp = (t == 0) ? t0v[n] : ET[n * T_ + t - 1];
    s[n] = (ET[n * T_ + t] - tp) * 0.5f;  // dt / RK
  }
}

// ---- relaxed top-K subset sampling + logprob (one block per sample) --------
__device__ __forceinline__ float blk_max(float v, float* sh) {
  int tid = threadIdx.x;
  sh[tid] = v; __syncthreads();
  for (int s = 128; s > 0; s >>= 1) {
    if (tid < s) sh[tid] = fmaxf(sh[tid], sh[tid + s]);
    __syncthreads();
  }
  float r = sh[0]; __syncthreads();
  return r;
}
__device__ __forceinline__ float blk_sum(float v, float* sh) {
  int tid = threadIdx.x;
  sh[tid] = v; __syncthreads();
  for (int s = 128; s > 0; s >>= 1) {
    if (tid < s) sh[tid] += sh[tid + s];
    __syncthreads();
  }
  float r = sh[0]; __syncthreads();
  return r;
}

__global__ __launch_bounds__(256) void ksubset_kernel(
    const float* __restrict__ logits_in, const float* __restrict__ gum,
    const float* __restrict__ mask, float* __restrict__ khot,
    float* __restrict__ logp, int t) {
  __shared__ float sh[256];
  const int n = blockIdx.x, tid = threadIdx.x;
  float lg[3], sel[3], sc[3];
#pragma unroll
  for (int j = 0; j < 3; ++j) {
    int idx = tid + j * 256;
    if (idx < L_) {
      float l0 = logits_in[n * L_ + idx];
      float m  = mask[n * L_ + idx];
      float lm = m * -1e6f + l0 * (1.0f - m);
      lg[j] = lm;
      sc[j] = lm + gum[((size_t)n * T_ + t) * L_ + idx];
    } else { lg[j] = -INFINITY; sc[j] = -INFINITY; }
    sel[j] = 0.0f;
  }
  for (int it = 0; it < KSUB; ++it) {
#pragma unroll
    for (int j = 0; j < 3; ++j)
      if (tid + j * 256 < L_)
        sc[j] += logf(fminf(fmaxf(1.0f - sel[j], 1e-6f), 1.0f));
    float mx = fmaxf(fmaxf(sc[0], sc[1]), sc[2]);
    mx = blk_max(mx, sh);
    float ex[3];
    float sm = 0.0f;
#pragma unroll
    for (int j = 0; j < 3; ++j) {
      ex[j] = (tid + j * 256 < L_) ? expf(sc[j] - mx) : 0.0f;
      sm += ex[j];
    }
    sm = blk_sum(sm, sh);
    float inv = 1.0f / sm;
#pragma unroll
    for (int j = 0; j < 3; ++j)
      if (tid + j * 256 < L_) sel[j] += ex[j] * inv;
  }
#pragma unroll
  for (int j = 0; j < 3; ++j) {
    int idx = tid + j * 256;
    if (idx < L_) {
      float kv = fminf(fmaxf(sel[j], 0.0f), 1.0f);
      khot[n * L_ + idx] = kv;
      sel[j] = kv;
    }
  }
  // logprob = sum khot * log_softmax(lg)
  float mx = fmaxf(fmaxf(lg[0], lg[1]), lg[2]);
  mx = blk_max(mx, sh);
  float sm = 0.0f;
#pragma unroll
  for (int j = 0; j < 3; ++j)
    if (tid + j * 256 < L_) sm += expf(lg[j] - mx);
  sm = blk_sum(sm, sh);
  float lse = logf(sm);
  float lp = 0.0f;
#pragma unroll
  for (int j = 0; j < 3; ++j)
    if (tid + j * 256 < L_) lp += sel[j] * (lg[j] - mx - lse);
  lp = blk_sum(lp, sh);
  if (tid == 0) logp[n * T_ + t] = lp;
}

// ---- ctrl scatter + adjacency einsum: tmp[n,i,:] = sum_j adj[n,i,j]*h[n,j,:]
__global__ __launch_bounds__(256) void adj_einsum_kernel(
    const float* __restrict__ khot, const float* __restrict__ adjs,
    const float* __restrict__ hf, __bf16* __restrict__ tmpb) {
  __shared__ float arow[32];
  const int ni = blockIdx.x;           // n*D + i
  const int n = ni / D_, i = ni - n * D_;
  const int tid = threadIdx.x;         // h column (H_==256)
  if (tid < D_) {
    int j = tid;
    float ctrl = 0.0f;
    if (j != i) {
      int p = i * (D_ - 1) + j - (j > i ? 1 : 0);  // off-diag flat index
      ctrl = khot[n * L_ + p];
    }
    arow[j] = adjs[i * D_ + j] * (1.0f - ctrl);
  }
  __syncthreads();
  const float* hn = hf + (size_t)n * DH_;
  float acc = 0.0f;
#pragma unroll
  for (int j = 0; j < D_; ++j) acc += arow[j] * hn[j * H_ + tid];
  tmpb[(size_t)ni * H_ + tid] = (__bf16)acc;
}

// ---- intensity head: softplus(i1 @ wi2 + bi2) ------------------------------
__global__ void inten_kernel(const __bf16* __restrict__ i1,
                             const float* __restrict__ wi2,
                             const float* __restrict__ bi2,
                             float* __restrict__ out, int t) {
  int r = blockIdx.x * 256 + threadIdx.x;  // ND_
  if (r >= ND_) return;
  float acc = bi2[0];
#pragma unroll 8
  for (int c = 0; c < HI_; ++c) acc += (float)i1[(size_t)r * HI_ + c] * wi2[c];
  int n = r / D_, d = r - n * D_;
  out[((size_t)n * T_ + t) * D_ + d] = (acc > 20.0f) ? acc : log1pf(expf(acc));
}

// ---------------------------------------------------------------------------
extern "C" void kernel_launch(void* const* d_in, const int* in_sizes, int n_in,
                              void* d_out, int out_size, void* d_ws,
                              size_t ws_size, hipStream_t stream) {
  (void)in_sizes; (void)n_in; (void)out_size; (void)ws_size;
  const float* ET   = (const float*)d_in[0];
  const float* ED   = (const float*)d_in[1];
  const float* T0   = (const float*)d_in[2];
  const float* MASK = (const float*)d_in[3];
  const float* GUM  = (const float*)d_in[4];
  const float* H0   = (const float*)d_in[5];
  const float* ADJS = (const float*)d_in[6];
  const float* WC1 = (const float*)d_in[7];  const float* BC1 = (const float*)d_in[8];
  const float* WC2 = (const float*)d_in[9];  const float* BC2 = (const float*)d_in[10];
  const float* WD  = (const float*)d_in[11]; const float* BD  = (const float*)d_in[12];
  const float* WO  = (const float*)d_in[13];
  const float* WI1 = (const float*)d_in[14]; const float* BI1 = (const float*)d_in[15];
  const float* WI2 = (const float*)d_in[16]; const float* BI2 = (const float*)d_in[17];
  const float* WR  = (const float*)d_in[18]; const float* BR  = (const float*)d_in[19];
  const float* WP  = (const float*)d_in[20]; const float* BP  = (const float*)d_in[21];
  const float* WM  = (const float*)d_in[22]; const float* BM  = (const float*)d_in[23];
  const float* WQ1 = (const float*)d_in[24]; const float* BQ1 = (const float*)d_in[25];
  const float* WQ2 = (const float*)d_in[26]; const float* BQ2 = (const float*)d_in[27];

  float* out_inten = (float*)d_out;              // (N,T,D)
  float* out_logp  = (float*)d_out + N_ * T_ * D_;

  // -- workspace carve-out (256B aligned) --
  char* base = (char*)d_ws; size_t off = 0;
  auto WS = [&](size_t bytes) -> void* {
    void* p = base + off;
    off = (off + bytes + 255) & ~(size_t)255;
    return p;
  };
  __bf16* wr_b  = (__bf16*)WS((size_t)DH_ * E_ * 2);
  __bf16* wp_b  = (__bf16*)WS((size_t)E_ * E_ * 2);
  __bf16* wm_b  = (__bf16*)WS((size_t)E_ * E_ * 2);
  __bf16* wq1_b = (__bf16*)WS((size_t)2 * E_ * HP_ * 2);
  __bf16* wq2_b = (__bf16*)WS((size_t)HP_ * L_ * 2);
  __bf16* wd_b  = (__bf16*)WS((size_t)H_ * H_ * 2);
  __bf16* wc1_b = (__bf16*)WS((size_t)H_ * HC_ * 2);
  __bf16* wc2_b = (__bf16*)WS((size_t)HC_ * H_ * 2);
  __bf16* wi1_b = (__bf16*)WS((size_t)H_ * HI_ * 2);
  float*  h_f   = (float*) WS((size_t)NDH_ * 4);
  __bf16* h_b   = (__bf16*)WS((size_t)NDH_ * 2);
  __bf16* e_b   = (__bf16*)WS((size_t)N_ * E_ * 2);
  __bf16* pm_b  = (__bf16*)WS((size_t)N_ * 2 * E_ * 2);
  __bf16* q1_b  = (__bf16*)WS((size_t)N_ * HP_ * 2);
  float*  lgts  = (float*) WS((size_t)N_ * L_ * 4);
  float*  khot  = (float*) WS((size_t)N_ * L_ * 4);
  __bf16* tmp_b = (__bf16*)WS((size_t)NDH_ * 2);
  __bf16* x_b   = (__bf16*)WS((size_t)NDH_ * 2);
  __bf16* g1_b  = (__bf16*)WS((size_t)ND_ * HC_ * 2);
  float*  k1    = (float*) WS((size_t)NDH_ * 4);
  float*  k2    = (float*) WS((size_t)NDH_ * 4);
  float*  k3    = (float*) WS((size_t)NDH_ * 4);
  __bf16* i1_b  = (__bf16*)WS((size_t)ND_ * HI_ * 2);
  float*  s_f   = (float*) WS((size_t)N_ * 4);

  auto conv = [&](const float* s, __bf16* d, int n) {
    convert_bf16_kernel<<<(n + 255) / 256, 256, 0, stream>>>(s, d, n);
  };
  conv(WR,  wr_b,  DH_ * E_);
  conv(WP,  wp_b,  E_ * E_);
  conv(WM,  wm_b,  E_ * E_);
  conv(WQ1, wq1_b, 2 * E_ * HP_);
  conv(WQ2, wq2_b, HP_ * L_);
  conv(WD,  wd_b,  H_ * H_);
  conv(WC1, wc1_b, H_ * HC_);
  conv(WC2, wc2_b, HC_ * H_);
  conv(WI1, wi1_b, H_ * HI_);

  hinit_kernel<<<NDH_ / 256, 256, 0, stream>>>(H0, h_f, h_b);

  // cfn first half: g1 = tanh(x @ wc1 + bc1)
  auto cfn1 = [&](const __bf16* xin) {
    gemm_bf16_kernel<4, 1, false, true>
        <<<dim3(ND_ / 64, HC_ / 64), 128, 0, stream>>>(
            xin, wc1_b, BC1, nullptr, g1_b, H_, HC_, HC_, 0);
  };

  for (int t = 0; t < T_; ++t) {
    // representation: e = tanh(h(N x DH) @ wr + br)
    gemm_bf16_kernel<4, 1, false, true><<<dim3(2, E_ / 64), 128, 0, stream>>>(
        h_b, wr_b, BR, nullptr, e_b, DH_, E_, E_, 0);
    // pos / mage into concatenated (N x 2E)
    gemm_bf16_kernel<4, 0, false, true><<<dim3(2, E_ / 64), 128, 0, stream>>>(
        e_b, wp_b, BP, nullptr, pm_b, E_, E_, 2 * E_, 0);
    gemm_bf16_kernel<4, 0, false, true><<<dim3(2, E_ / 64), 128, 0, stream>>>(
        e_b, wm_b, BM, nullptr, pm_b + E_, E_, E_, 2 * E_, 0);
    // policy MLP
    gemm_bf16_kernel<4, 1, false, true><<<dim3(2, HP_ / 64), 128, 0, stream>>>(
        pm_b, wq1_b, BQ1, nullptr, q1_b, 2 * E_, HP_, HP_, 0);
    // q2: 608 cols = 9 full 64-wide strips + one 32-wide remainder
    gemm_bf16_kernel<4, 0, true, false><<<dim3(2, L_ / 64), 128, 0, stream>>>(
        q1_b, wq2_b, BQ2, lgts, nullptr, HP_, L_, L_, 0);
    gemm_bf16_kernel<2, 0, true, false><<<dim3(2, 1), 128, 0, stream>>>(
        q1_b, wq2_b, BQ2, lgts, nullptr, HP_, L_, L_, (L_ / 64) * 64);
    // relaxed k-subset + logprob
    ksubset_kernel<<<N_, 256, 0, stream>>>(lgts, GUM, MASK, khot, out_logp, t);
    // control scatter + adjacency mix
    adj_einsum_kernel<<<ND_, 256, 0, stream>>>(khot, ADJS, h_f, tmp_b);
    // jump update: h += tanh(tmp@wd + obs*wo + bd)
    gemm_jump_kernel<<<dim3(ND_ / 64, H_ / 64), 128, 0, stream>>>(
        tmp_b, wd_b, BD, WO, ED, t, h_f, h_b);
    // ODE step scale s = dt/RK
    dt_kernel<<<1, 128, 0, stream>>>(ET, T0, s_f, t);
    // RK4 x 2, axpy/combine fused into the k-producing GEMM epilogues
    const dim3 gk(ND_ / 64, H_ / 64);
    for (int rk = 0; rk < 2; ++rk) {
      cfn1(h_b);
      gemm_cfn2_stage_kernel<<<gk, 128, 0, stream>>>(
          g1_b, wc2_b, BC2, h_f, s_f, 0.5f, k1, x_b);
      cfn1(x_b);
      gemm_cfn2_stage_kernel<<<gk, 128, 0, stream>>>(
          g1_b, wc2_b, BC2, h_f, s_f, 0.5f, k2, x_b);
      cfn1(x_b);
      gemm_cfn2_stage_kernel<<<gk, 128, 0, stream>>>(
          g1_b, wc2_b, BC2, h_f, s_f, 1.0f, k3, x_b);
      cfn1(x_b);
      gemm_cfn2_final_kernel<<<gk, 128, 0, stream>>>(
          g1_b, wc2_b, BC2, k1, k2, k3, s_f, h_f, h_b);
    }
    // intensity head
    gemm_bf16_kernel<4, 1, false, true>
        <<<dim3(ND_ / 64, HI_ / 64), 128, 0, stream>>>(
            h_b, wi1_b, BI1, nullptr, i1_b, H_, HI_, HI_, 0);
    inten_kernel<<<(ND_ + 255) / 256, 256, 0, stream>>>(i1_b, WI2, BI2,
                                                        out_inten, t);
  }
}
